// PerceiverAttention_15977278341576
// MI455X (gfx1250) — matrix operand order repaired and checked by reading, not compiled, not run on hardware
//
#include <hip/hip_runtime.h>
#include <hip/hip_bf16.h>

// MI455X / gfx1250, wave32. bf16 WMMA (v_wmma_f32_16x16x32_bf16) everywhere.
// Pipeline: weights->bf16T | LN->bf16 | GEMM(kv_x,+RoPE) | GEMM(kv_l) |
//           GEMM(q) | flash-attention | GEMM(out @ Wo) -> fp32.
// The big GEMM stages A-tiles into LDS with the Tensor Data Mover
// (tensor_load_to_lds + s_wait_tensorcnt), depth-3 prefetch over a 4-buffer
// LDS ring; B (weights, 2 MB bf16) streams straight from the 192 MB L2.

typedef __attribute__((ext_vector_type(16))) __bf16 v16bf;
typedef __attribute__((ext_vector_type(8)))  float  v8f;

#define LPAD 8320   // 8192 + 64 latents, padded to multiple of 128
#define LK   8256   // valid keys per (b,h)

#ifndef __has_builtin
#define __has_builtin(x) 0
#endif
#if defined(__HIP_DEVICE_COMPILE__) && \
    __has_builtin(__builtin_amdgcn_tensor_load_to_lds) && \
    __has_builtin(__builtin_amdgcn_s_wait_tensorcnt)
#define HAVE_TDM 1
#else
#define HAVE_TDM 0
#endif

struct U4x2 { uint4 a, b; };

static __device__ __forceinline__ v16bf frag_cast(uint4 a, uint4 b) {
  U4x2 t{a, b};
  return __builtin_bit_cast(v16bf, t);
}
static __device__ __forceinline__ unsigned short bf_bits(float f) {
  return __builtin_bit_cast(unsigned short, (__bf16)f);
}

#if HAVE_TDM
// Issue one TDM descriptor: load a 128-row x 32-col bf16 tile (8 KB), row
// stride Kd elements, from global `gsrc` into LDS byte offset `ldsOff`.
// D# layout per CDNA5 ISA 8.3/8.4 (2-D tensor; groups 2/3 zero).
static __device__ __forceinline__ void tdm_load_tile_a(
    const __bf16* gsrc, unsigned ldsOff, int Kd, int Mrows) {
  typedef __attribute__((ext_vector_type(4))) unsigned v4u_t;
  typedef __attribute__((ext_vector_type(8))) int      v8i_t;
  typedef __attribute__((ext_vector_type(4))) int      v4i_t;
  const unsigned long long ga = (unsigned long long)(size_t)(const void*)gsrc;
  v4u_t g0;
  g0[0] = 1u;                                   // count=1 (valid), user mode
  g0[1] = ldsOff;                               // lds_addr (bytes)
  g0[2] = (unsigned)ga;                         // global_addr[31:0]
  g0[3] = (unsigned)((ga >> 32) & 0x1ffffffull) // global_addr[56:32]
          | (2u << 30);                         // type=2 ("image")
  v8i_t g1;
  g1[0] = 1 << 16;                              // wg_mask=0, data_size=1 (2B)
  g1[1] = (Kd & 0xffff) << 16;                  // tensor_dim0[15:0]
  g1[2] = ((Kd >> 16) & 0xffff)                 // tensor_dim0[31:16]
          | ((Mrows & 0xffff) << 16);           // tensor_dim1[15:0]
  g1[3] = ((Mrows >> 16) & 0xffff)              // tensor_dim1[31:16]
          | (32 << 16);                         // tile_dim0 = 32 elems
  g1[4] = 128;                                  // tile_dim1 = 128, tile_dim2=0
  g1[5] = Kd;                                   // tensor_dim0_stride[31:0]
  g1[6] = 0;                                    // stride0 hi / stride1 lo
  g1[7] = 0;                                    // stride1 hi
  const v4i_t z4 = {0, 0, 0, 0};
#if __clang_major__ >= 23
  const v8i_t z8 = {0, 0, 0, 0, 0, 0, 0, 0};
  __builtin_amdgcn_tensor_load_to_lds(g0, g1, z4, z4, z8, 0);
#else
  __builtin_amdgcn_tensor_load_to_lds(g0, g1, z4, z4, 0);
#endif
}
#endif  // HAVE_TDM

// ---------------------------------------------------------------- transpose
// W is [Kd][Nd] fp32 row-major; WT is [Nd][Kd] bf16 row-major.
__global__ __launch_bounds__(256) void transpose_cvt_kernel(
    const float* __restrict__ W, __bf16* __restrict__ WT, int Kd, int Nd) {
  int idx = blockIdx.x * 256 + threadIdx.x;
  if (idx >= Kd * Nd) return;
  int n = idx / Kd;
  int k = idx - n * Kd;
  WT[idx] = (__bf16)W[(size_t)k * Nd + n];
}

// ---------------------------------------------------------------- layernorm
__global__ __launch_bounds__(256) void ln_rows_kernel(
    const float* __restrict__ X, const float* __restrict__ g,
    const float* __restrict__ be, __bf16* __restrict__ out) {
  const int row = blockIdx.x;
  const int t = threadIdx.x;
  const float4 xv = ((const float4*)(X + (size_t)row * 1024))[t];
  float s  = xv.x + xv.y + xv.z + xv.w;
  float sq = xv.x * xv.x + xv.y * xv.y + xv.z * xv.z + xv.w * xv.w;
#pragma unroll
  for (int m = 1; m < 32; m <<= 1) {
    s  += __shfl_xor(s,  m, 32);
    sq += __shfl_xor(sq, m, 32);
  }
  __shared__ float ss[8], sqs[8];
  const int wv = t >> 5;
  if ((t & 31) == 0) { ss[wv] = s; sqs[wv] = sq; }
  __syncthreads();
  float ts = 0.f, tq = 0.f;
#pragma unroll
  for (int w = 0; w < 8; ++w) { ts += ss[w]; tq += sqs[w]; }
  const float mu   = ts * (1.f / 1024.f);
  const float var  = tq * (1.f / 1024.f) - mu * mu;
  const float rstd = rsqrtf(var + 1e-5f);
  const float4 gv = ((const float4*)g)[t];
  const float4 bv = ((const float4*)be)[t];
  const unsigned u0 = bf_bits((xv.x - mu) * rstd * gv.x + bv.x);
  const unsigned u1 = bf_bits((xv.y - mu) * rstd * gv.y + bv.y);
  const unsigned u2 = bf_bits((xv.z - mu) * rstd * gv.z + bv.z);
  const unsigned u3 = bf_bits((xv.w - mu) * rstd * gv.w + bv.w);
  uint2 pk;
  pk.x = u0 | (u1 << 16);
  pk.y = u2 | (u3 << 16);
  ((uint2*)(out + (size_t)row * 1024))[t] = pk;
}

// ---------------------------------------------------------------- WMMA GEMM
// C[M,N] = A[M,Kd] * BT[N,Kd]^T  (A,BT bf16 row-major; accumulate f32).
// Workgroup tile 128x128; 8 waves 4(M) x 2(N); wave tile 32x64 = 2x4 WMMA
// accumulators. A staged in LDS via TDM ring; B streamed from L2.
enum { MODE_KVX = 0, MODE_KVL = 1, MODE_Q = 2, MODE_OUT = 3 };

template <int MODE>
__global__ __launch_bounds__(256) void gemm_wmma_kernel(
    const __bf16* __restrict__ A, const __bf16* __restrict__ BT, int Kd,
    __bf16* __restrict__ Kb, __bf16* __restrict__ Vb,
    __bf16* __restrict__ Qb, float* __restrict__ Ob,
    int rpbShift, int jOffset) {
  const int lane = threadIdx.x & 31;
  const int wv   = threadIdx.x >> 5;
  const int wM = wv >> 1, wN = wv & 1;
  const int l15 = lane & 15, hl = lane >> 4;
  const int m0 = blockIdx.y * 128, n0 = blockIdx.x * 128;

#if HAVE_TDM
  __shared__ alignas(16) __bf16 As[4][128 * 32];  // 32 KB ring, depth-3 PF
#else
  __shared__ alignas(16) __bf16 As[1][128 * 32];
#endif
  v8f acc[2][4] = {};

#if HAVE_TDM
  const int Mrows = (int)gridDim.y * 128;
  const __bf16* Abase = A + (size_t)m0 * Kd;
  if (wv == 0) {
#pragma unroll
    for (int p = 0; p < 3; ++p)
      if (p * 32 < Kd)
        tdm_load_tile_a(Abase + p * 32,
                        (unsigned)(size_t)(void*)&As[p][0], Kd, Mrows);
  }
#else
  const int aRow = threadIdx.x >> 1;
  const int aCol = (threadIdx.x & 1) * 16;
#endif

  int ib = 0;
  for (int kk = 0; kk < Kd; kk += 32, ++ib) {
#if HAVE_TDM
    const int cur = ib & 3;
    if (wv == 0) {
      const int rem = ((Kd - kk) >> 5) - 1;  // iterations after this one
      if (rem >= 3) {
        tdm_load_tile_a(Abase + kk + 96,
                        (unsigned)(size_t)(void*)&As[(ib + 3) & 3][0], Kd,
                        Mrows);
        __builtin_amdgcn_s_wait_tensorcnt((short)3);  // tile `ib` landed
      } else if (rem == 2) {
        __builtin_amdgcn_s_wait_tensorcnt((short)2);
      } else if (rem == 1) {
        __builtin_amdgcn_s_wait_tensorcnt((short)1);
      } else {
        __builtin_amdgcn_s_wait_tensorcnt((short)0);
      }
    }
    __syncthreads();  // publish tile `cur` to all waves
#else
    const int cur = 0;
    const uint4* asrc =
        (const uint4*)(A + (size_t)(m0 + aRow) * Kd + kk + aCol);
    uint4* adst = (uint4*)(As[0] + aRow * 32 + aCol);
    adst[0] = asrc[0];
    adst[1] = asrc[1];
    __syncthreads();
#endif

    v16bf aF[2];
#pragma unroll
    for (int mi = 0; mi < 2; ++mi) {
      // A layout: lane=row, K = 0..15 (lanes 0-15) / 16..31 (lanes 16-31)
      const uint4* p =
          (const uint4*)(As[cur] + (wM * 32 + mi * 16 + l15) * 32 + hl * 16);
      aF[mi] = frag_cast(p[0], p[1]);
    }
    v16bf bF[4];
#pragma unroll
    for (int ni = 0; ni < 4; ++ni) {
      // B layout: lane=col, K = 0..15 (lanes 0-15) / 16..31 (lanes 16-31)
      const int col = n0 + wN * 64 + ni * 16 + l15;
      const __bf16* bp = BT + (size_t)col * Kd + kk + hl * 16;
      __builtin_prefetch((const void*)(bp + 64), 0, 0);  // global_prefetch_b8
      const uint4* p = (const uint4*)bp;
      bF[ni] = frag_cast(p[0], p[1]);
    }
#pragma unroll
    for (int mi = 0; mi < 2; ++mi)
#pragma unroll
      for (int ni = 0; ni < 4; ++ni)
        acc[mi][ni] = __builtin_amdgcn_wmma_f32_16x16x32_bf16(
            false, aF[mi], false, bF[ni], (short)0, acc[mi][ni], false, false);
    __syncthreads();  // all waves done with tile `cur` before it is reused
  }

  const int colW = n0 + wN * 64;  // 64-aligned -> exactly one head
  const int rowW = m0 + wM * 32;

  if constexpr (MODE == MODE_OUT) {
    const int Nld = gridDim.x * 128;
#pragma unroll
    for (int mi = 0; mi < 2; ++mi) {
      const int rB = rowW + mi * 16 + hl * 8;
#pragma unroll
      for (int ni = 0; ni < 4; ++ni) {
        const int c = colW + ni * 16 + l15;
#pragma unroll
        for (int v = 0; v < 8; ++v)
          Ob[(size_t)(rB + v) * Nld + c] = acc[mi][ni][v];
      }
    }
  } else if constexpr (MODE == MODE_Q) {
    const int h = colW >> 6;
#pragma unroll
    for (int mi = 0; mi < 2; ++mi) {
      const int rB = rowW + mi * 16 + hl * 8;
      const int b  = rB >> 6;
      const int i0 = rB & 63;
#pragma unroll
      for (int ni = 0; ni < 4; ++ni) {
        const int d = ni * 16 + l15;
#pragma unroll
        for (int v = 0; v < 8; ++v)
          Qb[(((size_t)(b * 8 + h) * 64) + (i0 + v)) * 64 + d] =
              (__bf16)(acc[mi][ni][v] * 0.125f);  // d^-0.5 folded into q
      }
    }
  } else {  // KV modes
    const bool isK = colW < 512;
    const int  h   = isK ? (colW >> 6) : ((colW - 512) >> 6);
    const int  rpbMask = (1 << rpbShift) - 1;

    if (MODE == MODE_KVX && isK) {
      // fused RoPE: pairs (d, d+32) are accumulator subtiles (p, p+2)
#pragma unroll
      for (int p = 0; p < 2; ++p) {
        const int d = p * 16 + l15;  // 0..31
        const float fr = __powf(10000.f, -(float)d * (1.f / 32.f));
#pragma unroll
        for (int mi = 0; mi < 2; ++mi) {
          const int rB = rowW + mi * 16 + hl * 8;
          const int i0 = rB & rpbMask;
#pragma unroll
          for (int v = 0; v < 8; ++v) {
            float sv, cv;
            __sincosf((float)(i0 + v) * fr, &sv, &cv);
            const float lo = acc[mi][p][v];
            const float hi = acc[mi][p + 2][v];
            acc[mi][p][v]     = lo * cv - hi * sv;
            acc[mi][p + 2][v] = hi * cv + lo * sv;
          }
        }
      }
    }
#pragma unroll
    for (int mi = 0; mi < 2; ++mi) {
      const int rB = rowW + mi * 16 + hl * 8;
      const int b  = rB >> rpbShift;
      const int j0 = (rB & rpbMask) + jOffset;
      if (isK) {
        // K stored [bh][j][d] (row-major, B-friendly for q.kT)
#pragma unroll
        for (int ni = 0; ni < 4; ++ni) {
          const int d = ni * 16 + l15;
#pragma unroll
          for (int v = 0; v < 8; ++v)
            Kb[((size_t)(b * 8 + h) * LPAD + (j0 + v)) * 64 + d] =
                (__bf16)acc[mi][ni][v];
        }
      } else {
        // V stored transposed [bh][d][j]: per-lane j-contiguous -> b128 store
#pragma unroll
        for (int ni = 0; ni < 4; ++ni) {
          const int d = ni * 16 + l15;
          unsigned u[8];
#pragma unroll
          for (int v = 0; v < 8; ++v) u[v] = bf_bits(acc[mi][ni][v]);
          uint4 pk;
          pk.x = u[0] | (u[1] << 16);
          pk.y = u[2] | (u[3] << 16);
          pk.z = u[4] | (u[5] << 16);
          pk.w = u[6] | (u[7] << 16);
          *(uint4*)(Vb + ((size_t)(b * 8 + h) * 64 + d) * LPAD + j0) = pk;
        }
      }
    }
  }
}

// ------------------------------------------------------------ flash attention
__global__ __launch_bounds__(128) void attn_kernel(
    const __bf16* __restrict__ Qb, const __bf16* __restrict__ Kb,
    const __bf16* __restrict__ Vb, __bf16* __restrict__ Ob) {
  const int bh = blockIdx.x;
  const int lane = threadIdx.x & 31;
  const int wv = threadIdx.x >> 5;
  const int l15 = lane & 15, hl = lane >> 4;

  __shared__ alignas(16) __bf16 pbuf[4][16][32];

  v16bf aQ[2];
  {
    const int row = wv * 16 + l15;
#pragma unroll
    for (int c = 0; c < 2; ++c) {
      const uint4* p =
          (const uint4*)(Qb + ((size_t)bh * 64 + row) * 64 + c * 32 + hl * 16);
      aQ[c] = frag_cast(p[0], p[1]);
    }
  }

  float rmax[8], rsum[8];
#pragma unroll
  for (int v = 0; v < 8; ++v) { rmax[v] = -3.0e38f; rsum[v] = 0.f; }
  v8f accO[4] = {};

  for (int j0 = 0; j0 < LPAD; j0 += 32) {
    v8f s[2];
#pragma unroll
    for (int jt = 0; jt < 2; ++jt) {
      const int j = j0 + jt * 16 + l15;
      const __bf16* kp = Kb + ((size_t)bh * LPAD + j) * 64 + hl * 16;
      const uint4* p0 = (const uint4*)kp;
      const uint4* p1 = (const uint4*)(kp + 32);
      const v16bf b0 = frag_cast(p0[0], p0[1]);
      const v16bf b1 = frag_cast(p1[0], p1[1]);
      v8f z = {0.f, 0.f, 0.f, 0.f, 0.f, 0.f, 0.f, 0.f};
      z = __builtin_amdgcn_wmma_f32_16x16x32_bf16(false, aQ[0], false, b0,
                                                  (short)0, z, false, false);
      z = __builtin_amdgcn_wmma_f32_16x16x32_bf16(false, aQ[1], false, b1,
                                                  (short)0, z, false, false);
      if (j >= LK) {
#pragma unroll
        for (int v = 0; v < 8; ++v) z[v] = -3.0e38f;  // mask padded keys
      }
      s[jt] = z;
    }
    // online softmax update, 8 rows per lane (C layout: lane=col, vgpr=row)
#pragma unroll
    for (int v = 0; v < 8; ++v) {
      float mloc = fmaxf(s[0][v], s[1][v]);
#pragma unroll
      for (int m = 1; m < 16; m <<= 1)
        mloc = fmaxf(mloc, __shfl_xor(mloc, m, 32));
      const float mnew = fmaxf(rmax[v], mloc);
      const float corr = __expf(rmax[v] - mnew);
      rmax[v] = mnew;
      const float p0 = __expf(s[0][v] - mnew);
      const float p1 = __expf(s[1][v] - mnew);
      float ls = p0 + p1;
#pragma unroll
      for (int m = 1; m < 16; m <<= 1) ls += __shfl_xor(ls, m, 32);
      rsum[v] = rsum[v] * corr + ls;
#pragma unroll
      for (int sub = 0; sub < 4; ++sub) accO[sub][v] *= corr;
      const int prow = hl * 8 + v;
      pbuf[wv][prow][l15]      = (__bf16)p0;
      pbuf[wv][prow][16 + l15] = (__bf16)p1;
    }
    // reload P in A-fragment layout (wave-synchronous; compiler waits DScnt)
    v16bf pF;
    {
      const uint4* p = (const uint4*)(&pbuf[wv][l15][hl * 16]);
      pF = frag_cast(p[0], p[1]);
    }
#pragma unroll
    for (int sub = 0; sub < 4; ++sub) {
      const int d = sub * 16 + l15;
      const uint4* p =
          (const uint4*)(Vb + ((size_t)bh * 64 + d) * LPAD + j0 + hl * 16);
      const v16bf vF = frag_cast(p[0], p[1]);
      accO[sub] = __builtin_amdgcn_wmma_f32_16x16x32_bf16(
          false, pF, false, vF, (short)0, accO[sub], false, false);
    }
  }

  const int b = bh >> 3, h = bh & 7;
#pragma unroll
  for (int sub = 0; sub < 4; ++sub) {
    const int d = sub * 16 + l15;
#pragma unroll
    for (int v = 0; v < 8; ++v) {
      const int i = wv * 16 + hl * 8 + v;
      Ob[((size_t)(b * 64 + i)) * 512 + h * 64 + d] =
          (__bf16)(accO[sub][v] / rsum[v]);
    }
  }
}

// ---------------------------------------------------------------- launcher
extern "C" void kernel_launch(void* const* d_in, const int* in_sizes, int n_in,
                              void* d_out, int out_size, void* d_ws,
                              size_t ws_size, hipStream_t stream) {
  (void)in_sizes; (void)n_in; (void)out_size; (void)ws_size;
  const float* x   = (const float*)d_in[0];
  const float* lat = (const float*)d_in[1];
  const float* gx  = (const float*)d_in[2];
  const float* bx  = (const float*)d_in[3];
  const float* gl  = (const float*)d_in[4];
  const float* bl  = (const float*)d_in[5];
  const float* Wq  = (const float*)d_in[6];
  const float* Wkv = (const float*)d_in[7];
  const float* Wo  = (const float*)d_in[8];
  float* out = (float*)d_out;

  char* ws = (char*)d_ws;
  size_t off = 0;
  auto alloc = [&](size_t bytes) {
    char* p = ws + off;
    off = (off + bytes + 255) & ~(size_t)255;
    return p;
  };
  __bf16* xn   = (__bf16*)alloc((size_t)65536 * 1024 * 2);
  __bf16* lnl  = (__bf16*)alloc((size_t)512 * 1024 * 2);
  __bf16* WkvT = (__bf16*)alloc((size_t)1024 * 1024 * 2);
  __bf16* WqT  = (__bf16*)alloc((size_t)512 * 1024 * 2);
  __bf16* WoT  = (__bf16*)alloc((size_t)1024 * 512 * 2);
  __bf16* Qb   = (__bf16*)alloc((size_t)64 * 64 * 64 * 2);
  __bf16* Kbuf = (__bf16*)alloc((size_t)64 * LPAD * 64 * 2);
  __bf16* Vbuf = (__bf16*)alloc((size_t)64 * LPAD * 64 * 2);
  __bf16* Ab   = (__bf16*)alloc((size_t)512 * 512 * 2);

  transpose_cvt_kernel<<<(1024 * 1024 + 255) / 256, 256, 0, stream>>>(
      Wkv, WkvT, 1024, 1024);
  transpose_cvt_kernel<<<(1024 * 512 + 255) / 256, 256, 0, stream>>>(
      Wq, WqT, 1024, 512);
  transpose_cvt_kernel<<<(512 * 1024 + 255) / 256, 256, 0, stream>>>(
      Wo, WoT, 512, 1024);

  ln_rows_kernel<<<65536, 256, 0, stream>>>(x, gx, bx, xn);
  ln_rows_kernel<<<512, 256, 0, stream>>>(lat, gl, bl, lnl);

  // kv_x = xn @ Wkv  (+RoPE on k half), 65536x1024x1024
  gemm_wmma_kernel<MODE_KVX><<<dim3(8, 512), 256, 0, stream>>>(
      xn, WkvT, 1024, Kbuf, Vbuf, nullptr, nullptr, /*rpbShift=*/13,
      /*jOffset=*/0);
  // kv_l = ln @ Wkv, appended at j=8192
  gemm_wmma_kernel<MODE_KVL><<<dim3(8, 4), 256, 0, stream>>>(
      lnl, WkvT, 1024, Kbuf, Vbuf, nullptr, nullptr, /*rpbShift=*/6,
      /*jOffset=*/8192);
  // q = ln @ Wq (scaled)
  gemm_wmma_kernel<MODE_Q><<<dim3(4, 4), 256, 0, stream>>>(
      lnl, WqT, 1024, nullptr, nullptr, Qb, nullptr, /*rpbShift=*/6, 0);

  attn_kernel<<<64, 128, 0, stream>>>(Qb, Kbuf, Vbuf, Ab);

  // final projection -> fp32 output
  gemm_wmma_kernel<MODE_OUT><<<dim3(8, 4), 256, 0, stream>>>(
      Ab, WoT, 512, nullptr, nullptr, nullptr, out, /*rpbShift=*/6, 0);
}
